// TransformerLayer_82497731821807
// MI455X (gfx1250) — matrix operand order repaired and checked by reading, not compiled, not run on hardware
//
#include <hip/hip_runtime.h>
#include <hip/hip_bf16.h>
#include <math.h>

// ---------------- types / constants ----------------
typedef __bf16 bf16_t;
typedef __attribute__((ext_vector_type(16))) __bf16 v16bf;
typedef __attribute__((ext_vector_type(8)))  float  v8f;

#define D_MODEL 1024
#define N_HEAD  16
#define D_HEAD  64
#define D_CAT   1024
#define N_QKV   3072
#define HD_MLP  4096
#define SEQ_T   2048
#define BATCH   2
#define ROWS    (BATCH * SEQ_T)   // 4096
#define LN_EPS  1e-5f

union ABfrag { v16bf v; uint4 u[2]; };

// ---------------- fragment loaders ----------------
// A-fragment (16x32 bf16, M x K) from packed-A layout Ap[(k>>3)][m][k&7].
// Lane half=0: K 0..15 ; half=1: K 16..31 (contiguous, matching B).
__device__ __forceinline__ v16bf ldA_packed(const bf16_t* Ap, int M, int k0, int m, int half) {
  ABfrag f;
  f.u[0] = *(const uint4*)(Ap + ((size_t)((k0 >> 3) + 2 * half) * M + m) * 8);
  f.u[1] = *(const uint4*)(Ap + ((size_t)((k0 >> 3) + 2 * half + 1) * M + m) * 8);
  return f.v;
}
// A-fragment from a row-major row pointer.
__device__ __forceinline__ v16bf ldA_row(const bf16_t* rowbase, int k0, int half) {
  ABfrag f;
  f.u[0] = *(const uint4*)(rowbase + k0 + half * 16);
  f.u[1] = *(const uint4*)(rowbase + k0 + half * 16 + 8);
  return f.v;
}
// B-fragment (32x16 bf16, K x N) from packed-B layout Bp[(k>>4)][n][k&15].
// Lanes 0..15 hold K 0..15 of column n=lane; lanes 16..31 hold K 16..31.
__device__ __forceinline__ v16bf ldB_packed(const bf16_t* Bp, int N, int k0, int n, int half) {
  const bf16_t* p = Bp + ((size_t)((k0 >> 4) + half) * N + n) * 16;
  ABfrag f;
  f.u[0] = *(const uint4*)(p);
  f.u[1] = *(const uint4*)(p + 8);
  return f.v;
}
// B-fragment where this lane's 16 K-values are contiguous in memory.
__device__ __forceinline__ v16bf ldB_contig(const bf16_t* p) {
  ABfrag f;
  f.u[0] = *(const uint4*)(p);
  f.u[1] = *(const uint4*)(p + 8);
  return f.v;
}

__device__ __forceinline__ v8f wmma_bf16(v16bf a, v16bf b, v8f c) {
  // (neg_a, A, neg_b, B, c_mod, C, reuse_a, reuse_b)
  return __builtin_amdgcn_wmma_f32_16x16x32_bf16(false, a, false, b, (short)0, c, false, false);
}

// ---------------- shared GEMM mainloop ----------------
// Wave computes a 64x64 tile: c[4][4] accumulators, K stepped by 32.
// 16 b128 fragment loads per 16 WMMAs -> high WMMA issue density.
__device__ __forceinline__ void gemm_main(const bf16_t* Ap, const bf16_t* Bp,
                                          int M, int N, int K,
                                          int row0, int col0, int lane,
                                          v8f c[4][4]) {
  const int half = lane >> 4, l = lane & 15;
  for (int k0 = 0; k0 < K; k0 += 32) {
    if (k0 + 32 < K) {  // prefetch next B tile (global_prefetch_b8)
      __builtin_prefetch(Bp + ((size_t)(((k0 + 32) >> 4) + half) * N + col0 + l) * 16, 0, 3);
    }
    v16bf a[4], b[4];
#pragma unroll
    for (int i = 0; i < 4; ++i) a[i] = ldA_packed(Ap, M, k0, row0 + i * 16 + l, half);
#pragma unroll
    for (int j = 0; j < 4; ++j) b[j] = ldB_packed(Bp, N, k0, col0 + j * 16 + l, half);
#pragma unroll
    for (int i = 0; i < 4; ++i)
#pragma unroll
      for (int j = 0; j < 4; ++j)
        c[i][j] = wmma_bf16(a[i], b[j], c[i][j]);
  }
}

// Block geometry shared by all GEMM kernels: 8 waves = 2 (M) x 4 (N), tile 128x256.
__device__ __forceinline__ void gemm_coords(int& row0, int& col0, int& lane) {
  lane = threadIdx.x & 31;
  const int wave = threadIdx.x >> 5;
  row0 = blockIdx.y * 128 + (wave & 1) * 64;
  col0 = blockIdx.x * 256 + (wave >> 1) * 64;
}

// ---------------- weight pack: fp32 row-major -> bf16 packed-B ----------------
__global__ void pack_w_kernel(const float* __restrict__ W, bf16_t* __restrict__ Wp, int K, int N) {
  size_t idx = (size_t)blockIdx.x * 256 + threadIdx.x;
  if (idx >= (size_t)K * N) return;
  int n = (int)(idx % N);
  int k = (int)(idx / N);
  Wp[(size_t)(k >> 4) * N * 16 + (size_t)n * 16 + (k & 15)] = (bf16_t)W[idx];
}

// ---------------- LayerNorm + pack to A-fragment layout ----------------
__global__ __launch_bounds__(256) void ln_pack_kernel(const float* __restrict__ X,
                                                      const float* __restrict__ g,
                                                      const float* __restrict__ beta,
                                                      bf16_t* __restrict__ Zp) {
  const int row = blockIdx.x;
  const int tid = threadIdx.x;
  float4 x4 = ((const float4*)(X + (size_t)row * D_MODEL))[tid];
  float s  = x4.x + x4.y + x4.z + x4.w;
  float s2 = x4.x * x4.x + x4.y * x4.y + x4.z * x4.z + x4.w * x4.w;
  __shared__ float rs[256], rs2[256];
  rs[tid] = s; rs2[tid] = s2;
  __syncthreads();
  for (int off = 128; off > 0; off >>= 1) {
    if (tid < off) { rs[tid] += rs[tid + off]; rs2[tid] += rs2[tid + off]; }
    __syncthreads();
  }
  const float mu   = rs[0] * (1.0f / D_MODEL);
  const float var  = rs2[0] * (1.0f / D_MODEL) - mu * mu;
  const float rsig = rsqrtf(var + LN_EPS);
  float vals[4] = {x4.x, x4.y, x4.z, x4.w};
#pragma unroll
  for (int i = 0; i < 4; ++i) {
    int kf = tid * 4 + i;
    float z = (vals[i] - mu) * rsig * g[kf] + beta[kf];
    Zp[(size_t)(kf >> 3) * (ROWS * 8) + (size_t)row * 8 + (kf & 7)] = (bf16_t)z;
  }
}

// ---------------- GEMM 1: z @ w_qkv -> q, k (row-major), v (transposed) ----------------
__global__ __launch_bounds__(256) void gemm_qkv_kernel(const bf16_t* __restrict__ Ap,
                                                       const bf16_t* __restrict__ Bp,
                                                       bf16_t* __restrict__ qbuf,
                                                       bf16_t* __restrict__ kbuf,
                                                       bf16_t* __restrict__ vT) {
  int row0, col0, lane;
  gemm_coords(row0, col0, lane);
  v8f zero = {};
  v8f c[4][4];
#pragma unroll
  for (int i = 0; i < 4; ++i)
#pragma unroll
    for (int j = 0; j < 4; ++j) c[i][j] = zero;
  gemm_main(Ap, Bp, ROWS, N_QKV, D_MODEL, row0, col0, lane, c);
  const int half = lane >> 4, l = lane & 15;
#pragma unroll
  for (int i = 0; i < 4; ++i)
#pragma unroll
    for (int j = 0; j < 4; ++j)
#pragma unroll
      for (int r = 0; r < 8; ++r) {
        int m = row0 + i * 16 + r + 8 * half;
        int n = col0 + j * 16 + l;
        int head = n / 192, rem = n - head * 192;
        int sel = rem >> 6, dk = rem & 63;
        int b = m >> 11, t = m & (SEQ_T - 1);
        size_t bh = (size_t)(b * N_HEAD + head);
        bf16_t v = (bf16_t)c[i][j][r];
        if (sel == 0)       qbuf[(bh * SEQ_T + t) * 64 + dk] = v;
        else if (sel == 1)  kbuf[(bh * SEQ_T + t) * 64 + dk] = v;
        else                vT[(bh * 64 + dk) * SEQ_T + t] = v;
      }
}

// ---------------- Flash attention (one wave per 16-query block) ----------------
__global__ __launch_bounds__(256) void attn_kernel(const bf16_t* __restrict__ qbuf,
                                                   const bf16_t* __restrict__ kbuf,
                                                   const bf16_t* __restrict__ vT,
                                                   const int* __restrict__ mask,
                                                   bf16_t* __restrict__ attnP) {
  __shared__ __align__(16) bf16_t ptile[8][16 * 48];  // per-wave 16x32 P staging (stride 48)
  const int lane = threadIdx.x & 31, wave = threadIdx.x >> 5;
  const int half = lane >> 4, l = lane & 15;
  const int bh = blockIdx.x;
  const int bidx = bh >> 4, head = bh & (N_HEAD - 1);
  const int q0 = (blockIdx.y * 8 + wave) * 16;
  const bf16_t* qbase = qbuf + (size_t)bh * SEQ_T * 64;
  const bf16_t* kbase = kbuf + (size_t)bh * SEQ_T * 64;
  const bf16_t* vbase = vT + (size_t)bh * 64 * SEQ_T;
  const int* mrow = mask + (size_t)bidx * SEQ_T;
  bf16_t* pt = ptile[wave];

  const bf16_t* qrow = qbase + (size_t)(q0 + l) * 64;
  v16bf qa0 = ldA_row(qrow, 0, half);
  v16bf qa1 = ldA_row(qrow, 32, half);

  v8f zero = {};
  v8f o[4];
#pragma unroll
  for (int f = 0; f < 4; ++f) o[f] = zero;
  float mrun[8], lrun[8];
#pragma unroll
  for (int r = 0; r < 8; ++r) { mrun[r] = -3.0e38f; lrun[r] = 0.0f; }

  for (int j = 0; j < SEQ_T; j += 32) {
    // S = (Q @ K^T): two 16x16 tiles (keys j..j+15 and j+16..j+31)
    v8f s0 = zero, s1 = zero;
    {
      v16bf b0a = ldB_contig(kbase + (size_t)(j + l) * 64 + half * 16);
      v16bf b0b = ldB_contig(kbase + (size_t)(j + l) * 64 + 32 + half * 16);
      s0 = wmma_bf16(qa0, b0a, s0);
      s0 = wmma_bf16(qa1, b0b, s0);
      v16bf b1a = ldB_contig(kbase + (size_t)(j + 16 + l) * 64 + half * 16);
      v16bf b1b = ldB_contig(kbase + (size_t)(j + 16 + l) * 64 + 32 + half * 16);
      s1 = wmma_bf16(qa0, b1a, s1);
      s1 = wmma_bf16(qa1, b1b, s1);
    }
    const bool ok0 = mrow[j + l] != 0;
    const bool ok1 = mrow[j + 16 + l] != 0;
#pragma unroll
    for (int r = 0; r < 8; ++r) {
      float v0 = ok0 ? s0[r] * 0.125f : -3.0e38f;  // scale = D_HEAD^-0.5
      float v1 = ok1 ? s1[r] * 0.125f : -3.0e38f;
      float rm = fmaxf(v0, v1);
      rm = fmaxf(rm, __shfl_xor(rm, 1, 32));
      rm = fmaxf(rm, __shfl_xor(rm, 2, 32));
      rm = fmaxf(rm, __shfl_xor(rm, 4, 32));
      rm = fmaxf(rm, __shfl_xor(rm, 8, 32));
      float mn = fmaxf(mrun[r], rm);
      float corr = __expf(mrun[r] - mn);
      mrun[r] = mn;
      float p0 = __expf(v0 - mn);
      float p1 = __expf(v1 - mn);
      float rssum = p0 + p1;
      rssum += __shfl_xor(rssum, 1, 32);
      rssum += __shfl_xor(rssum, 2, 32);
      rssum += __shfl_xor(rssum, 4, 32);
      rssum += __shfl_xor(rssum, 8, 32);
      lrun[r] = lrun[r] * corr + rssum;
#pragma unroll
      for (int f = 0; f < 4; ++f) o[f][r] *= corr;
      bf16_t* prow = pt + (size_t)(r + 8 * half) * 48;  // C layout: row = r + 8*half, col = l
      prow[l] = (bf16_t)p0;
      prow[16 + l] = (bf16_t)p1;
    }
    asm volatile("s_wait_dscnt 0" ::: "memory");  // LDS stores visible before lane-transposed read
    v16bf pa;
    {
      ABfrag f;
      const bf16_t* pr = pt + (size_t)l * 48;
      f.u[0] = *(const uint4*)(pr + half * 16);
      f.u[1] = *(const uint4*)(pr + half * 16 + 8);
      pa = f.v;
    }
    // O += P @ V : V transposed so each lane's 16 keys are contiguous
#pragma unroll
    for (int f = 0; f < 4; ++f) {
      v16bf vb = ldB_contig(vbase + (size_t)(f * 16 + l) * SEQ_T + j + half * 16);
      o[f] = wmma_bf16(pa, vb, o[f]);
    }
  }
  // normalize and emit in packed-A layout feeding the tail GEMM
#pragma unroll
  for (int r = 0; r < 8; ++r) {
    float inv = 1.0f / lrun[r];
    int m = bidx * SEQ_T + q0 + r + 8 * half;
#pragma unroll
    for (int f = 0; f < 4; ++f) {
      int kf = head * 64 + f * 16 + l;
      attnP[(size_t)(kf >> 3) * (ROWS * 8) + (size_t)m * 8 + (kf & 7)] = (bf16_t)(o[f][r] * inv);
    }
  }
}

// ---------------- GEMM 2: attn @ w_tail + b_tail + x -> x1 (fp32) ----------------
__global__ __launch_bounds__(256) void gemm_tail_kernel(const bf16_t* __restrict__ Ap,
                                                        const bf16_t* __restrict__ Bp,
                                                        const float* __restrict__ bias,
                                                        const float* __restrict__ resid,
                                                        float* __restrict__ out) {
  int row0, col0, lane;
  gemm_coords(row0, col0, lane);
  v8f zero = {};
  v8f c[4][4];
#pragma unroll
  for (int i = 0; i < 4; ++i)
#pragma unroll
    for (int j = 0; j < 4; ++j) c[i][j] = zero;
  gemm_main(Ap, Bp, ROWS, D_MODEL, D_CAT, row0, col0, lane, c);
  const int half = lane >> 4, l = lane & 15;
#pragma unroll
  for (int i = 0; i < 4; ++i)
#pragma unroll
    for (int j = 0; j < 4; ++j)
#pragma unroll
      for (int r = 0; r < 8; ++r) {
        int m = row0 + i * 16 + r + 8 * half;
        int n = col0 + j * 16 + l;
        size_t idx = (size_t)m * D_MODEL + n;
        out[idx] = c[i][j][r] + bias[n] + resid[idx];
      }
}

// ---------------- GEMM 3: z2 @ w1 + b1, exact GELU -> h (packed-A bf16) ----------------
__global__ __launch_bounds__(256) void gemm_mlp1_kernel(const bf16_t* __restrict__ Ap,
                                                        const bf16_t* __restrict__ Bp,
                                                        const float* __restrict__ bias,
                                                        bf16_t* __restrict__ hP) {
  int row0, col0, lane;
  gemm_coords(row0, col0, lane);
  v8f zero = {};
  v8f c[4][4];
#pragma unroll
  for (int i = 0; i < 4; ++i)
#pragma unroll
    for (int j = 0; j < 4; ++j) c[i][j] = zero;
  gemm_main(Ap, Bp, ROWS, HD_MLP, D_MODEL, row0, col0, lane, c);
  const int half = lane >> 4, l = lane & 15;
#pragma unroll
  for (int i = 0; i < 4; ++i)
#pragma unroll
    for (int j = 0; j < 4; ++j)
#pragma unroll
      for (int r = 0; r < 8; ++r) {
        int m = row0 + i * 16 + r + 8 * half;
        int n = col0 + j * 16 + l;
        float v = c[i][j][r] + bias[n];
        float gel = 0.5f * v * (1.0f + erff(v * 0.70710678118654752f));  // exact erf GELU
        hP[(size_t)(n >> 3) * (ROWS * 8) + (size_t)m * 8 + (n & 7)] = (bf16_t)gel;
      }
}

// ---------------- GEMM 4: h @ w2 + b2 + x1 -> out (fp32) ----------------
__global__ __launch_bounds__(256) void gemm_mlp2_kernel(const bf16_t* __restrict__ Ap,
                                                        const bf16_t* __restrict__ Bp,
                                                        const float* __restrict__ bias,
                                                        const float* __restrict__ resid,
                                                        float* __restrict__ out) {
  int row0, col0, lane;
  gemm_coords(row0, col0, lane);
  v8f zero = {};
  v8f c[4][4];
#pragma unroll
  for (int i = 0; i < 4; ++i)
#pragma unroll
    for (int j = 0; j < 4; ++j) c[i][j] = zero;
  gemm_main(Ap, Bp, ROWS, D_MODEL, HD_MLP, row0, col0, lane, c);
  const int half = lane >> 4, l = lane & 15;
#pragma unroll
  for (int i = 0; i < 4; ++i)
#pragma unroll
    for (int j = 0; j < 4; ++j)
#pragma unroll
      for (int r = 0; r < 8; ++r) {
        int m = row0 + i * 16 + r + 8 * half;
        int n = col0 + j * 16 + l;
        size_t idx = (size_t)m * D_MODEL + n;
        out[idx] = c[i][j][r] + bias[n] + resid[idx];
      }
}

// ---------------- launch ----------------
extern "C" void kernel_launch(void* const* d_in, const int* in_sizes, int n_in,
                              void* d_out, int out_size, void* d_ws, size_t ws_size,
                              hipStream_t stream) {
  (void)in_sizes; (void)n_in; (void)out_size; (void)ws_size;
  const float* x      = (const float*)d_in[0];
  const int*   mask   = (const int*)  d_in[1];
  const float* w_qkv  = (const float*)d_in[2];
  const float* w_tail = (const float*)d_in[3];
  const float* b_tail = (const float*)d_in[4];
  const float* ln1_g  = (const float*)d_in[5];
  const float* ln1_b  = (const float*)d_in[6];
  const float* ln2_g  = (const float*)d_in[7];
  const float* ln2_b  = (const float*)d_in[8];
  const float* w1     = (const float*)d_in[9];
  const float* b1     = (const float*)d_in[10];
  const float* w2     = (const float*)d_in[11];
  const float* b2     = (const float*)d_in[12];
  float* out = (float*)d_out;

  char* ws = (char*)d_ws;
  size_t off = 0;
  auto alloc = [&](size_t bytes) -> char* {
    char* p = ws + off;
    off += (bytes + 255) & ~(size_t)255;
    return p;
  };
  bf16_t* wqkvp  = (bf16_t*)alloc((size_t)D_MODEL * N_QKV * 2);
  bf16_t* wtailp = (bf16_t*)alloc((size_t)D_CAT * D_MODEL * 2);
  bf16_t* w1p    = (bf16_t*)alloc((size_t)D_MODEL * HD_MLP * 2);
  bf16_t* w2p    = (bf16_t*)alloc((size_t)HD_MLP * D_MODEL * 2);
  bf16_t* z1p    = (bf16_t*)alloc((size_t)ROWS * D_MODEL * 2);
  bf16_t* qb     = (bf16_t*)alloc((size_t)ROWS * D_CAT * 2);
  bf16_t* kb     = (bf16_t*)alloc((size_t)ROWS * D_CAT * 2);
  bf16_t* vtb    = (bf16_t*)alloc((size_t)ROWS * D_CAT * 2);
  bf16_t* attnp  = (bf16_t*)alloc((size_t)ROWS * D_CAT * 2);
  float*  x1     = (float*) alloc((size_t)ROWS * D_MODEL * 4);
  bf16_t* z2p    = (bf16_t*)alloc((size_t)ROWS * D_MODEL * 2);
  bf16_t* hp     = (bf16_t*)alloc((size_t)ROWS * HD_MLP * 2);

  dim3 blk(256);
  pack_w_kernel<<<(D_MODEL * N_QKV) / 256, blk, 0, stream>>>(w_qkv, wqkvp, D_MODEL, N_QKV);
  pack_w_kernel<<<(D_CAT * D_MODEL) / 256, blk, 0, stream>>>(w_tail, wtailp, D_CAT, D_MODEL);
  pack_w_kernel<<<(D_MODEL * HD_MLP) / 256, blk, 0, stream>>>(w1, w1p, D_MODEL, HD_MLP);
  pack_w_kernel<<<(HD_MLP * D_MODEL) / 256, blk, 0, stream>>>(w2, w2p, HD_MLP, D_MODEL);

  ln_pack_kernel<<<ROWS, blk, 0, stream>>>(x, ln1_g, ln1_b, z1p);
  gemm_qkv_kernel<<<dim3(N_QKV / 256, ROWS / 128), blk, 0, stream>>>(z1p, wqkvp, qb, kb, vtb);
  attn_kernel<<<dim3(BATCH * N_HEAD, SEQ_T / 128), blk, 0, stream>>>(qb, kb, vtb, mask, attnp);
  gemm_tail_kernel<<<dim3(D_MODEL / 256, ROWS / 128), blk, 0, stream>>>(attnp, wtailp, b_tail, x, x1);
  ln_pack_kernel<<<ROWS, blk, 0, stream>>>(x1, ln2_g, ln2_b, z2p);
  gemm_mlp1_kernel<<<dim3(HD_MLP / 256, ROWS / 128), blk, 0, stream>>>(z2p, w1p, b1, hp);
  gemm_mlp2_kernel<<<dim3(D_MODEL / 256, ROWS / 128), blk, 0, stream>>>(hp, w2p, b2, x1, out);
}
